// SelfAttention_37220186587176
// MI455X (gfx1250) — matrix operand order repaired and checked by reading, not compiled, not run on hardware
//
#include <hip/hip_runtime.h>
#include <hip/hip_bf16.h>

typedef __bf16  v16bf __attribute__((ext_vector_type(16)));
typedef __bf16  v8bf  __attribute__((ext_vector_type(8)));
typedef __bf16  v4bf  __attribute__((ext_vector_type(4)));
typedef float   v8f   __attribute__((ext_vector_type(8)));
typedef float   v4fl  __attribute__((ext_vector_type(4)));

// B=8, L=1024, D=1024, H=16, hd=64  ->  M = B*L = 8192, K = N = 1024
#define GM 8192
#define GK 1024
#define GN 1024

// ---------------------------------------------------------------- helpers

__device__ __forceinline__ v8f wmma_bf16(v16bf a, v16bf b, v8f c) {
  return __builtin_amdgcn_wmma_f32_16x16x32_bf16(
      /*neg_a=*/false, a, /*neg_b=*/false, b,
      /*c_mod=*/(short)0, c, /*reuse_a=*/false, /*reuse_b=*/false);
}

// A-matrix 16x32 bf16 fragment (ISA 7.12.2): lane<16 holds row (lane&15),
// K = {0..7, 16..23}; lane>=16 same row, K = {8..15, 24..31}.
__device__ __forceinline__ v16bf fragA(const __bf16* rowp, int hi, int k0) {
  v16bf r;
  *(v8bf*)&r        = *(const v8bf*)(rowp + k0 + hi * 8);
  *((v8bf*)&r + 1)  = *(const v8bf*)(rowp + k0 + 16 + hi * 8);
  return r;
}

// B-matrix 32x16 bf16 fragment: lane holds column (lane&15); lanes 0-15 get
// K = {0..7, 16..23}, lanes 16-31 get K = {8..15, 24..31} (matches fragA).
__device__ __forceinline__ v16bf fragB(const __bf16* colp, int hi, int k0) {
  v16bf r;
  *(v8bf*)&r        = *(const v8bf*)(colp + k0 + hi * 8);
  *((v8bf*)&r + 1)  = *(const v8bf*)(colp + k0 + 16 + hi * 8);
  return r;
}

// Low 32 bits of a flat pointer to LDS == wave-relative LDS byte offset.
__device__ __forceinline__ unsigned lds_off(const void* p) {
  return (unsigned)(unsigned long long)p;
}

// CDNA5 async DMA: global -> LDS, 16 bytes per lane, tracked by ASYNCcnt.
__device__ __forceinline__ void async_load_b128(unsigned lds_dst,
                                                const void* gsrc) {
  asm volatile("global_load_async_to_lds_b128 %0, %1, off"
               :: "v"(lds_dst), "v"(gsrc) : "memory");
}

// Force a wave-uniform condition into an SGPR so the compiler emits a scalar
// branch (s_cbranch) instead of EXEC masking. Critical around WMMA: the ISA
// requires EXEC == all-1s for WMMA, and WMMA is NOT skipped when EXEC == 0.
__device__ __forceinline__ bool uniform(bool c) {
  return __builtin_amdgcn_readfirstlane((int)c) != 0;
}

// ---------------------------------------------------------------- fp32 -> bf16

__global__ __launch_bounds__(256) void cvt_bf16_kernel(
    const float* __restrict__ in, __bf16* __restrict__ out, int n4) {
  int idx = blockIdx.x * 256 + threadIdx.x;
  if (idx >= n4) return;
  v4fl v = ((const v4fl*)in)[idx];
  v4bf o;
#pragma unroll
  for (int e = 0; e < 4; ++e) o[e] = (__bf16)v[e];
  ((v4bf*)out)[idx] = o;
}

// W [K][N] fp32 -> Wt [N][K] bf16 (column-major W = row-major Wt)
__global__ __launch_bounds__(256) void transpose_bf16_kernel(
    const float* __restrict__ W, __bf16* __restrict__ Wt) {
  int idx = blockIdx.x * 256 + threadIdx.x;   // < 1024*1024
  int k = idx >> 10, n = idx & 1023;
  Wt[(size_t)n * GK + k] = (__bf16)W[(size_t)k * GN + n];
}

// ---------------------------------------------------------------- GEMM
// C[M,N] = A[M,K] * Bt[N,K]^T + bias, bf16 in, f32 accumulate.
// Macro tile 128x256, 8 waves (2 x 4), wave tile 64x64, K-step 32,
// double-buffered LDS filled by global_load_async_to_lds_b128.
// OMODE 0: fp32 row-major [M][N]      (final projection -> d_out)
// OMODE 1: bf16 head-major [B,H,L,hd] (Q, K projections)
// OMODE 2: bf16 [B,H,hd,L]            (V projection, pre-transposed for attn)
template <int OMODE>
__global__ __launch_bounds__(256) void gemm256_kernel(
    const __bf16* __restrict__ A, const __bf16* __restrict__ Bt,
    const float* __restrict__ bias, void* __restrict__ outp) {
  __shared__ alignas(16) __bf16 As[2][128 * 32];
  __shared__ alignas(16) __bf16 Bs[2][256 * 32];

  const int tid  = threadIdx.x;
  const int wave = tid >> 5;
  const int lane = tid & 31;
  const int ln   = lane & 15;
  const int hi   = lane >> 4;
  const int wm   = wave >> 2;  // 0..1
  const int wn   = wave & 3;   // 0..3
  const size_t m0 = (size_t)blockIdx.y * 128;
  const size_t n0 = (size_t)blockIdx.x * 256;

  v8f acc[4][4];
#pragma unroll
  for (int i = 0; i < 4; ++i)
#pragma unroll
    for (int j = 0; j < 4; ++j)
#pragma unroll
      for (int e = 0; e < 8; ++e) acc[i][j][e] = 0.0f;

  // 6 async 16B chunks per thread per K-step (2 for A, 4 for B)
  auto issue_tile = [&](int buf, int k0) {
#pragma unroll
    for (int i = 0; i < 2; ++i) {
      int c = tid + i * 256;             // 0..511
      int row = c >> 2, col = (c & 3) * 8;
      async_load_b128(lds_off(&As[buf][row * 32 + col]),
                      &A[(m0 + row) * GK + k0 + col]);
    }
#pragma unroll
    for (int i = 0; i < 4; ++i) {
      int c = tid + i * 256;             // 0..1023
      int row = c >> 2, col = (c & 3) * 8;
      async_load_b128(lds_off(&Bs[buf][row * 32 + col]),
                      &Bt[(n0 + row) * GK + k0 + col]);
    }
  };

  issue_tile(0, 0);
#pragma unroll 2
  for (int k0 = 0, it = 0; k0 < GK; k0 += 32, ++it) {
    const int cur = it & 1;
    if (k0 + 32 < GK) {
      issue_tile(cur ^ 1, k0 + 32);
      // 6 newer (next-tile) asyncs may stay in flight; in-order completion
      // guarantees the current tile has landed.
      asm volatile("s_wait_asynccnt 0x6" ::: "memory");
    } else {
      asm volatile("s_wait_asynccnt 0x0" ::: "memory");
    }
    __syncthreads();

    v16bf bfr[4];
#pragma unroll
    for (int f = 0; f < 4; ++f)
      bfr[f] = fragB(&Bs[cur][(wn * 64 + f * 16 + ln) * 32], hi, 0);
#pragma unroll
    for (int mi = 0; mi < 4; ++mi) {
      v16bf afr = fragA(&As[cur][(wm * 64 + mi * 16 + ln) * 32], hi, 0);
#pragma unroll
      for (int f = 0; f < 4; ++f)
        acc[mi][f] = wmma_bf16(afr, bfr[f], acc[mi][f]);
    }
    __syncthreads();  // all reads of buf done before it is refilled
  }

  // epilogue — C layout: VGPR j, lanes<16 -> M=j, lanes>=16 -> M=8+j, N=ln
#pragma unroll
  for (int mi = 0; mi < 4; ++mi) {
#pragma unroll
    for (int ni = 0; ni < 4; ++ni) {
      const int n    = (int)n0 + wn * 64 + ni * 16 + ln;
      const float bv = bias[n];
#pragma unroll
      for (int j = 0; j < 8; ++j) {
        const int m   = (int)m0 + wm * 64 + mi * 16 + j + hi * 8;
        const float v = acc[mi][ni][j] + bv;
        if constexpr (OMODE == 0) {
          ((float*)outp)[(size_t)m * GN + n] = v;
        } else {
          const int b = m >> 10, l = m & 1023, h = n >> 6, d = n & 63;
          if constexpr (OMODE == 1)
            ((__bf16*)outp)[(((size_t)b * 16 + h) * 1024 + l) * 64 + d] = (__bf16)v;
          else
            ((__bf16*)outp)[(((size_t)b * 16 + h) * 64 + d) * 1024 + l] = (__bf16)v;
        }
      }
    }
  }
}

// ---------------------------------------------------------------- RoPE (in place, head-major [B,H,L,64])

__global__ __launch_bounds__(256) void rope_kernel(
    __bf16* __restrict__ qh, __bf16* __restrict__ kh) {
  int idx = blockIdx.x * 256 + threadIdx.x;  // < B*H*L*32 = 4194304
  int i  = idx & 31;
  int l  = (idx >> 5) & 1023;
  int bh = idx >> 15;
  float invf = __powf(10000.0f, -(float)(2 * i) / 64.0f);
  float ang  = (float)l * invf;
  float s, c;
  __sincosf(ang, &s, &c);
  size_t base = (((size_t)bh * 1024) + l) * 64 + 2 * i;
  float x1 = (float)qh[base], x2 = (float)qh[base + 1];
  qh[base]     = (__bf16)(x1 * c - x2 * s);
  qh[base + 1] = (__bf16)(x1 * s + x2 * c);
  x1 = (float)kh[base];  x2 = (float)kh[base + 1];
  kh[base]     = (__bf16)(x1 * c - x2 * s);
  kh[base + 1] = (__bf16)(x1 * s + x2 * c);
}

// ---------------------------------------------------------------- flash attention
// grid (L/128, H, B), 256 threads. Wave w owns query rows [qtile + 16w, +16).
// qh,kh: [B,H,L,64] bf16; vT: [B,H,64,L] bf16; attn out: [B,L,D] bf16.
// K/V tiles double-buffered in LDS, filled with async-to-LDS DMA.
__global__ __launch_bounds__(256) void flash_attn_kernel(
    const __bf16* __restrict__ qh, const __bf16* __restrict__ kh,
    const __bf16* __restrict__ vT, __bf16* __restrict__ attn) {
  __shared__ alignas(16) __bf16 Ks[2][64 * 64];   // [key][hd]
  __shared__ alignas(16) __bf16 Vs[2][64 * 64];   // [hd][key]
  __shared__ alignas(16) __bf16 Ps[8 * 16 * 64];  // per-wave P staging

  const int tid  = threadIdx.x;
  const int wave = tid >> 5;
  const int lane = tid & 31;
  const int ln   = lane & 15;
  const int hi   = lane >> 4;
  const int bh   = blockIdx.z * 16 + blockIdx.y;
  const int qr   = blockIdx.x * 128 + wave * 16;

  const __bf16* qb = qh + (size_t)bh * 1024 * 64;
  const __bf16* kb = kh + (size_t)bh * 1024 * 64;
  const __bf16* vb = vT + (size_t)bh * 64 * 1024;

  // 4 async 16B chunks per thread per 64-key tile (2 for K, 2 for V)
  auto issue_kv = [&](int buf, int k0) {
#pragma unroll
    for (int i = 0; i < 2; ++i) {
      int c = tid + i * 256;             // 0..511
      int row = c >> 3, col = (c & 7) * 8;
      async_load_b128(lds_off(&Ks[buf][row * 64 + col]),
                      &kb[(size_t)(k0 + row) * 64 + col]);
      async_load_b128(lds_off(&Vs[buf][row * 64 + col]),
                      &vb[(size_t)row * 1024 + k0 + col]);
    }
  };

  // Q fragments straight from global (A layout), kept in registers
  v16bf qf[2];
#pragma unroll
  for (int s = 0; s < 2; ++s)
    qf[s] = fragA(qb + (size_t)(qr + ln) * 64, hi, s * 32);

  float m[8], lsum[8];
  v8f   o[4];
#pragma unroll
  for (int j = 0; j < 8; ++j) { m[j] = -1e30f; lsum[j] = 0.0f; }
#pragma unroll
  for (int f = 0; f < 4; ++f)
#pragma unroll
    for (int e = 0; e < 8; ++e) o[f][e] = 0.0f;

  const int ktiles = (blockIdx.x + 1) * 2;  // causal: keys <= qtile end
  issue_kv(0, 0);
#pragma unroll 2
  for (int kt = 0; kt < ktiles; ++kt) {
    const int k0  = kt * 64;
    const int cur = kt & 1;
    if (kt + 1 < ktiles) {
      issue_kv(cur ^ 1, k0 + 64);
      asm volatile("s_wait_asynccnt 0x4" ::: "memory");
    } else {
      asm volatile("s_wait_asynccnt 0x0" ::: "memory");
    }
    __syncthreads();

    // Wave-uniform causal tile classification (scalar branches via
    // readfirstlane so EXEC stays all-1s around every WMMA):
    //   k0 >  qr+15 : tile fully masked for this wave -> skip all compute
    //   k0+63 <= qr : tile fully unmasked -> skip per-element masking
    if (uniform(k0 <= qr + 15)) {
      // S = Q * K^T  (16 rows x 64 keys, 4 N-frags x 2 K-steps)
      v8f sf[4];
#pragma unroll
      for (int f = 0; f < 4; ++f)
#pragma unroll
        for (int e = 0; e < 8; ++e) sf[f][e] = 0.0f;
#pragma unroll
      for (int s = 0; s < 2; ++s)
#pragma unroll
        for (int f = 0; f < 4; ++f) {
          v16bf kf = fragB(&Ks[cur][(f * 16 + ln) * 64], hi, s * 32);
          sf[f] = wmma_bf16(qf[s], kf, sf[f]);
        }

      // online softmax (row j lives in VGPR index j / j+8, cols across lanes)
      const bool needMask = uniform((k0 + 63) > qr);  // diagonal tile
#pragma unroll
      for (int j = 0; j < 8; ++j) {
        const int row = qr + j + hi * 8;
        float mx = -1e30f;
        if (needMask) {
#pragma unroll
          for (int f = 0; f < 4; ++f) {
            const int col = k0 + f * 16 + ln;
            float v = sf[f][j] * 0.125f;       // 1/sqrt(hd)
            v = (col <= row) ? v : -1e30f;     // causal mask
            sf[f][j] = v;
            mx = fmaxf(mx, v);
          }
        } else {
#pragma unroll
          for (int f = 0; f < 4; ++f) {
            float v = sf[f][j] * 0.125f;
            sf[f][j] = v;
            mx = fmaxf(mx, v);
          }
        }
#pragma unroll
        for (int off = 1; off < 16; off <<= 1)
          mx = fmaxf(mx, __shfl_xor(mx, off, 32));
        const float nm   = fmaxf(m[j], mx);
        const float corr = __expf(m[j] - nm);
        float rs = 0.0f;
#pragma unroll
        for (int f = 0; f < 4; ++f) {
          const float p = __expf(sf[f][j] - nm);
          sf[f][j] = p;
          rs += p;
        }
#pragma unroll
        for (int off = 1; off < 16; off <<= 1) rs += __shfl_xor(rs, off, 32);
        lsum[j] = lsum[j] * corr + rs;
        m[j]    = nm;
#pragma unroll
        for (int f = 0; f < 4; ++f) o[f][j] *= corr;
      }

      // restage P: C layout -> A layout via per-wave LDS
      __bf16* pp = &Ps[wave * 1024];
#pragma unroll
      for (int j = 0; j < 8; ++j)
#pragma unroll
        for (int f = 0; f < 4; ++f)
          pp[(j + hi * 8) * 64 + f * 16 + ln] = (__bf16)sf[f][j];
      asm volatile("s_wait_dscnt 0x0" ::: "memory");  // cross-lane LDS dep

      // O += P * V
#pragma unroll
      for (int s = 0; s < 2; ++s) {
        v16bf pf = fragA(pp + (size_t)ln * 64, hi, s * 32);
#pragma unroll
        for (int f = 0; f < 4; ++f) {
          v16bf vf = fragB(&Vs[cur][(f * 16 + ln) * 64], hi, s * 32);
          o[f] = wmma_bf16(pf, vf, o[f]);
        }
      }
    }
    __syncthreads();  // all reads of buf done before it is refilled
  }

  // normalize and write attn output [B, L, D] bf16
  const int b = blockIdx.z, h = blockIdx.y;
#pragma unroll
  for (int j = 0; j < 8; ++j) {
    const int row   = qr + j + hi * 8;
    const float inv = 1.0f / lsum[j];
#pragma unroll
    for (int f = 0; f < 4; ++f) {
      const int d = f * 16 + ln;
      attn[((size_t)b * 1024 + row) * 1024 + h * 64 + d] =
          (__bf16)(o[f][j] * inv);
    }
  }
}

// ---------------------------------------------------------------- launch

extern "C" void kernel_launch(void* const* d_in, const int* in_sizes, int n_in,
                              void* d_out, int out_size, void* d_ws,
                              size_t ws_size, hipStream_t stream) {
  (void)in_sizes; (void)n_in; (void)out_size; (void)ws_size;

  const float* x  = (const float*)d_in[0];
  const float* Wq = (const float*)d_in[1];
  const float* bq = (const float*)d_in[2];
  const float* Wk = (const float*)d_in[3];
  const float* bk = (const float*)d_in[4];
  const float* Wv = (const float*)d_in[5];
  const float* bv = (const float*)d_in[6];
  const float* Wo = (const float*)d_in[7];
  const float* bo = (const float*)d_in[8];
  float* out = (float*)d_out;

  char* ws = (char*)d_ws;
  __bf16* xb   = (__bf16*)(ws + 0);          // 16 MB  x bf16 [M,K]
  __bf16* wqt  = (__bf16*)(ws + 16777216);   //  2 MB  Wq^T bf16
  __bf16* wkt  = (__bf16*)(ws + 18874368);   //  2 MB
  __bf16* wvt  = (__bf16*)(ws + 20971520);   //  2 MB
  __bf16* wot  = (__bf16*)(ws + 23068672);   //  2 MB
  __bf16* qhp  = (__bf16*)(ws + 25165824);   // 16 MB  q [B,H,L,64]
  __bf16* khp  = (__bf16*)(ws + 41943040);   // 16 MB  k [B,H,L,64]
  __bf16* vtp  = (__bf16*)(ws + 58720256);   // 16 MB  v [B,H,64,L]
  __bf16* attn = (__bf16*)(ws + 75497472);   // 16 MB  attn out [B,L,D]

  cvt_bf16_kernel<<<8192, 256, 0, stream>>>(x, xb, 2097152);  // 8M / 4
  transpose_bf16_kernel<<<4096, 256, 0, stream>>>(Wq, wqt);
  transpose_bf16_kernel<<<4096, 256, 0, stream>>>(Wk, wkt);
  transpose_bf16_kernel<<<4096, 256, 0, stream>>>(Wv, wvt);
  transpose_bf16_kernel<<<4096, 256, 0, stream>>>(Wo, wot);

  dim3 gg(GN / 256, GM / 128);  // (4, 64)
  gemm256_kernel<1><<<gg, 256, 0, stream>>>(xb, wqt, bq, qhp);
  gemm256_kernel<1><<<gg, 256, 0, stream>>>(xb, wkt, bk, khp);
  gemm256_kernel<2><<<gg, 256, 0, stream>>>(xb, wvt, bv, vtp);

  rope_kernel<<<16384, 256, 0, stream>>>(qhp, khp);  // 4M pairs

  flash_attn_kernel<<<dim3(8, 16, 8), 256, 0, stream>>>(qhp, khp, vtp, attn);

  gemm256_kernel<0><<<gg, 256, 0, stream>>>(attn, wot, bo, out);
}
